// SWCATransformer_45251775430664
// MI455X (gfx1250) — matrix operand order repaired and checked by reading, not compiled, not run on hardware
//
#include <hip/hip_runtime.h>

#define DEV __device__ __forceinline__

typedef __attribute__((ext_vector_type(16))) __bf16 v16bf;
typedef __attribute__((ext_vector_type(8)))  __bf16 v8bf;
typedef __attribute__((ext_vector_type(2)))  __bf16 v2bf;
typedef __attribute__((ext_vector_type(8)))  float  v8f;

static constexpr int CB   = 16;
static constexpr int CC   = 128;
static constexpr int CH   = 96;
static constexpr int CW   = 96;
static constexpr int CHW  = CH * CW;        // 9216
static constexpr int CHD  = 32;             // head dim
static constexpr int CNW  = 12;             // windows per row
static constexpr int CNWIN = 144;
static constexpr int NTOK = CB * CHW;       // 147456
static constexpr float CSCALE = 0.17677669529663687f;  // 1/sqrt(32)

// ---------- small helpers ----------
DEV float gelu_exact(float x) {
    return 0.5f * x * (1.0f + erff(x * 0.70710678118654752f));
}

DEV v8f wmma_bf16(v16bf a, v16bf b, v8f c) {
    return __builtin_amdgcn_wmma_f32_16x16x32_bf16(false, a, false, b, (short)0, c,
                                                   false, false);
}

// A fragment: 16x32 bf16, rows m0..m0+15, K cols k0..k0+31, src row-major, ld mult of 8.
// Per lane: two contiguous 16B runs -> 2x (ds|global)_load_b128.
DEV v16bf load_a_frag(const __bf16* src, int ld, int m0, int k0, int lane) {
    int m = lane & 15, hi = lane >> 4;
    const __bf16* p = src + (m0 + m) * ld + k0 + hi * 8;
    v8bf lo = *(const v8bf*)(p);
    v8bf hh = *(const v8bf*)(p + 16);
    return __builtin_shufflevector(lo, hh, 0, 1, 2, 3, 4, 5, 6, 7,
                                           8, 9, 10, 11, 12, 13, 14, 15);
}

// B fragment for logical B[k][n] with src stored [n][k] row-major (ld = K stride).
// Per lane: two contiguous 16B runs -> 2x (ds|global)_load_b128.
DEV v16bf load_bt_frag(const __bf16* src, int ld, int n0, int k0, int lane) {
    int n = lane & 15, kh = lane >> 4;
    const __bf16* p = src + (n0 + n) * ld + k0 + kh * 8;
    v8bf lo = *(const v8bf*)(p);
    v8bf hh = *(const v8bf*)(p + 16);
    return __builtin_shufflevector(lo, hh, 0, 1, 2, 3, 4, 5, 6, 7,
                                           8, 9, 10, 11, 12, 13, 14, 15);
}

// ---------- kernels ----------
__global__ void zero_kernel(float* p) {
    if (threadIdx.x == 0 && blockIdx.x == 0) *p = 0.0f;
}

__global__ void cvt_bf16_kernel(const float* __restrict__ in,
                                __bf16* __restrict__ out, int n) {
    int i = blockIdx.x * 256 + threadIdx.x;
    if (i < n) out[i] = (__bf16)in[i];
}

// LayerNorm over the channel dim, image layout (B,C,H,W). One thread per token.
__global__ void ln_kernel(const float* __restrict__ in,
                          const float* __restrict__ scale,
                          const float* __restrict__ bias,
                          float* __restrict__ out) {
    int t = blockIdx.x * 128 + threadIdx.x;      // token id in [0, NTOK)
    int b = t / CHW, yx = t % CHW;
    const float* p = in + (size_t)b * CC * CHW + yx;
    float s = 0.f, s2 = 0.f;
#pragma unroll 4
    for (int c = 0; c < CC; ++c) {
        float v = p[c * CHW];
        s += v; s2 += v * v;
    }
    float mu  = s * (1.0f / CC);
    float var = s2 * (1.0f / CC) - mu * mu;
    float inv = rsqrtf(var + 1e-5f);
    float* q = out + (size_t)b * CC * CHW + yx;
#pragma unroll 4
    for (int c = 0; c < CC; ++c) {
        float v = p[c * CHW];
        q[c * CHW] = (v - mu) * inv * scale[c] + bias[c];
    }
}

__global__ void add3_kernel(const float* a, const float* b, const float* c,
                            float* o, int n) {
    int i = blockIdx.x * 256 + threadIdx.x;
    if (i < n) o[i] = a[i] + b[i] + c[i];
}

__global__ void add2_kernel(const float* a, const float* b, float* o, int n) {
    int i = blockIdx.x * 256 + threadIdx.x;
    if (i < n) o[i] = a[i] + b[i];
}

// Fused window attention. grid = B*NH*NWIN, block = 128 (4 waves).
__global__ void attn_kernel(const float* __restrict__ xq_img,
                            const float* __restrict__ xk_img,
                            const float* __restrict__ xv_img,
                            const float* __restrict__ pe,     // 15x15
                            const float* __restrict__ wq, const float* __restrict__ bq,
                            const float* __restrict__ wk, const float* __restrict__ bk,
                            const float* __restrict__ wv, const float* __restrict__ bv,
                            const float* __restrict__ wo, const float* __restrict__ bo,
                            int shift,
                            float* __restrict__ out_img,
                            float* __restrict__ counter) {
    __shared__ alignas(16) __bf16 XQ[64 * 32], XK[64 * 32], XV[64 * 32];
    // weight matrices kept in natural [out][in] order = [n][k] for B-operand use
    __shared__ alignas(16) __bf16 WQ[32 * 32], WK[32 * 32], WV[32 * 32], WO[32 * 32];
    __shared__ float  BQ[32], BK[32], BV[32], BO[32];
    __shared__ float  PE[225];
    __shared__ alignas(16) __bf16 QS[64 * 32];   // [tok][d]  (A operand)
    __shared__ alignas(16) __bf16 KS[64 * 32];   // [tok][d]  ([n][k] for QK^T)
    __shared__ alignas(16) __bf16 VT[32 * 64];   // [d][tok]  ([n][k] for A@V)
    __shared__ alignas(16) __bf16 AS[64 * 64];   // [p][q]    (A operand)
    __shared__ alignas(16) __bf16 OS[64 * 32];   // [p][d]    (A operand)

    int bid  = blockIdx.x;
    int win  = bid % CNWIN;
    int head = (bid / CNWIN) % 4;
    int b    = bid / (CNWIN * 4);
    int wy = win / CNW, wx = win % CNW;
    int tid = threadIdx.x, lane = tid & 31, wave = tid >> 5;

    // gather window tokens, token-major so consecutive lanes hit consecutive ix
#pragma unroll
    for (int i = 0; i < 16; ++i) {
        int idx = i * 128 + tid;
        int d = idx >> 6, p = idx & 63;
        int iy = wy * 8 + (p >> 3), ix = wx * 8 + (p & 7);
        if (shift) { iy = (iy + 4) % CH; ix = (ix + 4) % CW; }
        size_t g = ((size_t)(b * CC + head * CHD + d)) * CHW + iy * CW + ix;
        XQ[p * 32 + d] = (__bf16)xq_img[g];
        XK[p * 32 + d] = (__bf16)xk_img[g];
        XV[p * 32 + d] = (__bf16)xv_img[g];
    }
    // weights copied in natural order (already [n][k])
#pragma unroll
    for (int i = 0; i < 8; ++i) {
        int idx = i * 128 + tid;
        WQ[idx] = (__bf16)wq[idx];
        WK[idx] = (__bf16)wk[idx];
        WV[idx] = (__bf16)wv[idx];
        WO[idx] = (__bf16)wo[idx];
    }
    {
        int t = tid;
        if      (t < 32)  BQ[t]       = bq[t];
        else if (t < 64)  BK[t - 32]  = bk[t - 32];
        else if (t < 96)  BV[t - 64]  = bv[t - 64];
        else              BO[t - 96]  = bo[t - 96];
    }
    for (int i = tid; i < 225; i += 128) PE[i] = pe[i];
    __syncthreads();

    int m0 = wave * 16;
    int n = lane & 15, hi = lane >> 4;

    // ---- Q/K/V projections: 6 independent WMMA chains issued back-to-back ----
    {
        v16bf aq = load_a_frag(XQ, 32, m0, 0, lane);
        v16bf ak = load_a_frag(XK, 32, m0, 0, lane);
        v16bf av = load_a_frag(XV, 32, m0, 0, lane);
        v16bf bq0 = load_bt_frag(WQ, 32, 0,  0, lane);
        v16bf bq1 = load_bt_frag(WQ, 32, 16, 0, lane);
        v16bf bk0 = load_bt_frag(WK, 32, 0,  0, lane);
        v16bf bk1 = load_bt_frag(WK, 32, 16, 0, lane);
        v16bf bv0 = load_bt_frag(WV, 32, 0,  0, lane);
        v16bf bv1 = load_bt_frag(WV, 32, 16, 0, lane);
        v8f cq0 = {}, cq1 = {}, ck0 = {}, ck1 = {}, cv0 = {}, cv1 = {};
        cq0 = wmma_bf16(aq, bq0, cq0);
        cq1 = wmma_bf16(aq, bq1, cq1);
        ck0 = wmma_bf16(ak, bk0, ck0);
        ck1 = wmma_bf16(ak, bk1, ck1);
        cv0 = wmma_bf16(av, bv0, cv0);
        cv1 = wmma_bf16(av, bv1, cv1);
        float bi_q0 = BQ[n], bi_q1 = BQ[16 + n];
        float bi_k0 = BK[n], bi_k1 = BK[16 + n];
        float bi_v0 = BV[n], bi_v1 = BV[16 + n];
#pragma unroll
        for (int r = 0; r < 8; ++r) {
            int m = m0 + r + 8 * hi;
            QS[m * 32 + n]      = (__bf16)(cq0[r] + bi_q0);
            QS[m * 32 + 16 + n] = (__bf16)(cq1[r] + bi_q1);
            KS[m * 32 + n]      = (__bf16)(ck0[r] + bi_k0);
            KS[m * 32 + 16 + n] = (__bf16)(ck1[r] + bi_k1);
        }
        // V stored transposed [d][tok]: pairs (r,r+1) adjacent -> b32 stores
        __bf16* v0base = VT + n * 64 + m0 + 8 * hi;
        __bf16* v1base = VT + (16 + n) * 64 + m0 + 8 * hi;
#pragma unroll
        for (int r = 0; r < 8; r += 2) {
            v2bf p0, p1;
            p0[0] = (__bf16)(cv0[r] + bi_v0); p0[1] = (__bf16)(cv0[r + 1] + bi_v0);
            p1[0] = (__bf16)(cv1[r] + bi_v1); p1[1] = (__bf16)(cv1[r + 1] + bi_v1);
            *(v2bf*)(v0base + r) = p0;
            *(v2bf*)(v1base + r) = p1;
        }
    }
    __syncthreads();

    // scores S = Q K^T * scale + pe + mask  (wave owns rows m0..m0+15, all 4 n-tiles)
    v8f s[4];
    {
        v16bf aq = load_a_frag(QS, 32, m0, 0, lane);
#pragma unroll
        for (int nt = 0; nt < 4; ++nt) {
            v16bf bk_ = load_bt_frag(KS, 32, nt * 16, 0, lane);
            v8f c = {};
            s[nt] = wmma_bf16(aq, bk_, c);
        }
    }
    bool mrow = shift && (win >= CNWIN - CNW);
    bool mcol = shift && ((win % CNW) == CNW - 1);
#pragma unroll
    for (int nt = 0; nt < 4; ++nt) {
        int q = nt * 16 + n;
        int q0 = q >> 3, q1 = q & 7;
#pragma unroll
        for (int r = 0; r < 8; ++r) {
            int p = m0 + r + 8 * hi;
            int p0 = p >> 3, p1 = p & 7;
            float v = s[nt][r] * CSCALE + PE[(q0 - p0 + 7) * 15 + (q1 - p1 + 7)];
            if (mrow && ((p >= 32) != (q >= 32))) v = -1e30f;
            if (mcol && ((p1 >= 4) != (q1 >= 4))) v = -1e30f;
            s[nt][r] = v;
        }
    }
    // softmax across each row (64 values = 4 n-tiles x 16 lanes of the half-wave)
    float cnt_part = 0.f;
#pragma unroll
    for (int r = 0; r < 8; ++r) {
        float mx = fmaxf(fmaxf(s[0][r], s[1][r]), fmaxf(s[2][r], s[3][r]));
        for (int off = 1; off < 16; off <<= 1)
            mx = fmaxf(mx, __shfl_xor(mx, off, 32));
        float e[4], sum = 0.f;
#pragma unroll
        for (int nt = 0; nt < 4; ++nt) { e[nt] = __expf(s[nt][r] - mx); sum += e[nt]; }
        for (int off = 1; off < 16; off <<= 1)
            sum += __shfl_xor(sum, off, 32);
        float inv = 1.0f / sum;
        int p = m0 + r + 8 * hi;
#pragma unroll
        for (int nt = 0; nt < 4; ++nt) {
            float a = e[nt] * inv;
            cnt_part += a * a;
            AS[p * 64 + nt * 16 + n] = (__bf16)a;
        }
    }
    for (int off = 1; off < 32; off <<= 1)
        cnt_part += __shfl_xor(cnt_part, off, 32);
    if (lane == 0) atomicAdd(counter, cnt_part);
    __syncthreads();

    // O = A @ V   (K = 64 tokens -> 2 chained K-steps, two interleaved accumulators)
    {
        v8f o0 = {}, o1 = {};
#pragma unroll
        for (int kt = 0; kt < 2; ++kt) {
            v16bf aa = load_a_frag(AS, 64, m0, kt * 32, lane);
            v16bf b0 = load_bt_frag(VT, 64, 0,  kt * 32, lane);
            v16bf b1 = load_bt_frag(VT, 64, 16, kt * 32, lane);
            o0 = wmma_bf16(aa, b0, o0);
            o1 = wmma_bf16(aa, b1, o1);
        }
#pragma unroll
        for (int r = 0; r < 8; ++r) {
            int m = m0 + r + 8 * hi;
            OS[m * 32 + n]      = (__bf16)o0[r];
            OS[m * 32 + 16 + n] = (__bf16)o1[r];
        }
    }
    __syncthreads();

    // Y = O @ Wo^T + bo, scatter back to image (undoing the roll for shifted blocks).
    // Both WMMAs issued before either epilogue; pairs (r,r+1) -> float2 stores
    // (wrap-safe: pairs start at even px, +4 roll never splits a pair at the seam).
    {
        v16bf ao  = load_a_frag(OS, 32, m0, 0, lane);
        v16bf bw0 = load_bt_frag(WO, 32, 0,  0, lane);
        v16bf bw1 = load_bt_frag(WO, 32, 16, 0, lane);
        v8f c0 = {}, c1 = {};
        c0 = wmma_bf16(ao, bw0, c0);
        c1 = wmma_bf16(ao, bw1, c1);
        float bb0 = BO[n], bb1 = BO[16 + n];
        size_t base0 = ((size_t)(b * CC + head * CHD + n)) * CHW;
        size_t base1 = ((size_t)(b * CC + head * CHD + 16 + n)) * CHW;
#pragma unroll
        for (int r = 0; r < 8; r += 2) {
            int p  = m0 + r + 8 * hi;               // even
            int iy = wy * 8 + (p >> 3), ix = wx * 8 + (p & 7);
            if (shift) { iy = (iy + 4) % CH; ix = (ix + 4) % CW; }
            size_t off = (size_t)iy * CW + ix;
            *(float2*)(out_img + base0 + off) = make_float2(c0[r] + bb0, c0[r + 1] + bb0);
            *(float2*)(out_img + base1 + off) = make_float2(c1[r] + bb1, c1[r + 1] + bb1);
        }
    }
}

// Fused 2-layer MLP with exact GELU. grid = NTOK/32, block = 256 (8 waves, 32 tokens).
// Weights are pre-converted bf16 in [n][k] order -> B fragments are two contiguous 16B runs per lane.
__global__ void mlp_kernel(const float* __restrict__ xin,
                           const __bf16* __restrict__ w1, const float* __restrict__ b1,
                           const __bf16* __restrict__ w2, const float* __restrict__ b2,
                           float* __restrict__ out) {
    __shared__ alignas(16) __bf16 AT[32 * 128];   // input tokens (bf16)
    __shared__ alignas(16) __bf16 H1[32 * 512];   // hidden activations (bf16)

    int tid = threadIdx.x, lane = tid & 31, w = tid >> 5;
    int t0 = blockIdx.x * 32;

    // stage input tile: A[tok][c] = img[b][c][y][x]
#pragma unroll
    for (int i = 0; i < 16; ++i) {
        int idx = i * 256 + tid;
        int tok = idx & 31, c = idx >> 5;
        int t = t0 + tok, b = t / CHW, yx = t % CHW;
        AT[tok * 128 + c] = (__bf16)xin[((size_t)(b * CC + c)) * CHW + yx];
    }
    __syncthreads();

    int mslab = w & 1, grp = w >> 1;     // grp in [0,4)
    int m0 = mslab * 16;
    int n = lane & 15, hi = lane >> 4;

    // phase 1: H1 = gelu(A @ W1^T + b1); pairs of n-tiles share one A-fragment
    v16bf afr[4];
#pragma unroll
    for (int kt = 0; kt < 4; ++kt)
        afr[kt] = load_a_frag(AT, 128, m0, kt * 32, lane);
    for (int nt = 0; nt < 8; nt += 2) {
        int n0 = grp * 128 + nt * 16;
        __builtin_prefetch(w1 + (size_t)(n0 + 32) * 128, 0, 0);
        v8f a0 = {}, a1 = {};
#pragma unroll
        for (int kt = 0; kt < 4; ++kt) {
            v16bf b0  = load_bt_frag(w1, 128, n0,      kt * 32, lane);
            v16bf b1f = load_bt_frag(w1, 128, n0 + 16, kt * 32, lane);
            a0 = wmma_bf16(afr[kt], b0, a0);
            a1 = wmma_bf16(afr[kt], b1f, a1);
        }
        float bias0 = b1[n0 + n], bias1 = b1[n0 + 16 + n];
#pragma unroll
        for (int r = 0; r < 8; ++r) {
            int m = m0 + r + 8 * hi;
            H1[m * 512 + n0 + n]      = (__bf16)gelu_exact(a0[r] + bias0);
            H1[m * 512 + n0 + 16 + n] = (__bf16)gelu_exact(a1[r] + bias1);
        }
    }
    __syncthreads();

    // phase 2: out = gelu(H1 @ W2^T + b2); both of this wave's n-tiles in one sweep.
    // Adjacent r are adjacent tokens (same image row run) -> float2 stores.
    {
        int n0 = grp * 32;
        v8f a0 = {}, a1 = {};
        for (int kt = 0; kt < 16; ++kt) {
            v16bf a   = load_a_frag(H1, 512, m0, kt * 32, lane);
            v16bf b0  = load_bt_frag(w2, 512, n0,      kt * 32, lane);
            v16bf b1f = load_bt_frag(w2, 512, n0 + 16, kt * 32, lane);
            a0 = wmma_bf16(a, b0, a0);
            a1 = wmma_bf16(a, b1f, a1);
        }
        float bias0 = b2[n0 + n], bias1 = b2[n0 + 16 + n];
#pragma unroll
        for (int r = 0; r < 8; r += 2) {
            int t = t0 + m0 + r + 8 * hi;               // even, pair stays in-bounds
            int b = t / CHW, yx = t % CHW;
            float2 v0 = make_float2(gelu_exact(a0[r] + bias0),
                                    gelu_exact(a0[r + 1] + bias0));
            float2 v1 = make_float2(gelu_exact(a1[r] + bias1),
                                    gelu_exact(a1[r + 1] + bias1));
            *(float2*)(out + ((size_t)(b * CC + n0 + n)) * CHW + yx)      = v0;
            *(float2*)(out + ((size_t)(b * CC + n0 + 16 + n)) * CHW + yx) = v1;
        }
    }
}

// ---------- driver ----------
extern "C" void kernel_launch(void* const* d_in, const int* in_sizes, int n_in,
                              void* d_out, int out_size, void* d_ws, size_t ws_size,
                              hipStream_t stream) {
    (void)in_sizes; (void)n_in; (void)ws_size;
    const float* q_feat  = (const float*)d_in[0];
    const float* k_feat  = (const float*)d_in[1];
    const float* v_feat  = (const float*)d_in[2];
    const float* ln_q_s  = (const float*)d_in[3];
    const float* ln_q_b  = (const float*)d_in[4];
    const float* ln_k_s  = (const float*)d_in[5];
    const float* ln_k_b  = (const float*)d_in[6];
    const float* ln_v_s  = (const float*)d_in[7];
    const float* ln_v_b  = (const float*)d_in[8];
    const float* ln_sw_s = (const float*)d_in[9];
    const float* ln_sw_b = (const float*)d_in[10];
    const float* msa_pe  = (const float*)d_in[11];  // (1,2,15,15)
    const float* msa_wq  = (const float*)d_in[12];  // (1,2,32,32)
    const float* msa_bq  = (const float*)d_in[13];  // (1,2,32)
    const float* msa_wk  = (const float*)d_in[14];
    const float* msa_bk  = (const float*)d_in[15];
    const float* msa_wv  = (const float*)d_in[16];
    const float* msa_bv  = (const float*)d_in[17];
    const float* msa_wo  = (const float*)d_in[18];
    const float* msa_bo  = (const float*)d_in[19];
    const float* fc1_w   = (const float*)d_in[20];  // (512,128)
    const float* fc1_b   = (const float*)d_in[21];
    const float* fc2_w   = (const float*)d_in[22];  // (128,512)
    const float* fc2_b   = (const float*)d_in[23];

    const size_t IMG = (size_t)CB * CC * CHW;       // 18,874,368
    const int WELT = 512 * 128;                     // 65536 weights per FC layer
    float* ws    = (float*)d_ws;
    float* nq    = ws + 0 * IMG;
    float* nk    = ws + 1 * IMG;
    float* nv    = ws + 2 * IMG;   // later reused for sw_v, then sw_msa
    float* att   = ws + 3 * IMG;
    float* wmsa  = ws + 4 * IMG;
    float* swv   = nv;
    float* swmsa = nv;
    __bf16* w1bf = (__bf16*)(ws + 5 * IMG);
    __bf16* w2bf = w1bf + WELT;

    float* counter = (float*)d_out + (out_size - 1);
    float* out_img = (float*)d_out;

    const int ln_grid  = NTOK / 128;                // 1152
    const int ew_grid  = (int)((IMG + 255) / 256);
    const int att_grid = CB * 4 * CNWIN;            // 9216
    const int mlp_grid = NTOK / 32;                 // 4608
    const int cvt_grid = (WELT + 255) / 256;        // 256

    zero_kernel<<<1, 32, 0, stream>>>(counter);
    cvt_bf16_kernel<<<cvt_grid, 256, 0, stream>>>(fc1_w, w1bf, WELT);
    cvt_bf16_kernel<<<cvt_grid, 256, 0, stream>>>(fc2_w, w2bf, WELT);

    ln_kernel<<<ln_grid, 128, 0, stream>>>(q_feat, ln_q_s, ln_q_b, nq);
    ln_kernel<<<ln_grid, 128, 0, stream>>>(k_feat, ln_k_s, ln_k_b, nk);
    ln_kernel<<<ln_grid, 128, 0, stream>>>(v_feat, ln_v_s, ln_v_b, nv);

    // W-MSA (no shift)
    attn_kernel<<<att_grid, 128, 0, stream>>>(
        nq, nk, nv, msa_pe,
        msa_wq, msa_bq, msa_wk, msa_bk, msa_wv, msa_bv, msa_wo, msa_bo,
        0, att, counter);

    add3_kernel<<<ew_grid, 256, 0, stream>>>(att, q_feat, k_feat, wmsa, (int)IMG);

    ln_kernel<<<ln_grid, 128, 0, stream>>>(wmsa, ln_sw_s, ln_sw_b, swv);

    // SW-MSA (shifted), second set of weights
    attn_kernel<<<att_grid, 128, 0, stream>>>(
        nq, nk, swv, msa_pe + 225,
        msa_wq + 1024, msa_bq + 32, msa_wk + 1024, msa_bk + 32,
        msa_wv + 1024, msa_bv + 32, msa_wo + 1024, msa_bo + 32,
        1, att, counter);

    add2_kernel<<<ew_grid, 256, 0, stream>>>(att, wmsa, swmsa, (int)IMG);

    mlp_kernel<<<mlp_grid, 256, 0, stream>>>(swmsa, w1bf, fc1_b, w2bf, fc2_b, out_img);
}
